// Morpher_69638599737737
// MI455X (gfx1250) — matrix operand order repaired and checked by reading, not compiled, not run on hardware
//
#include <hip/hip_runtime.h>

// Problem constants (match reference)
#define B_    4
#define T_    1024
#define N_    8
#define K_    4
#define D_    64
#define H_    32          // N_*K_
#define OD_   192         // 3*D
#define QTILE 128

typedef __attribute__((ext_vector_type(16))) __bf16        v16bf;
typedef __attribute__((ext_vector_type(8)))  float         v8f;
typedef __attribute__((ext_vector_type(8)))  unsigned int  v8u;

__device__ __forceinline__ unsigned short f2bf(float f) {
  __bf16 h = (__bf16)f;
  return __builtin_bit_cast(unsigned short, h);
}

// u32-index offset inside a 16-bit A fragment row: pairs (2v,2v+1) live at
// d = (v<4 ? 2v : 16+2(v-4)) (+8*half, +32*kstep)  ->  u32 idx = v<4 ? v : v+4
__device__ __forceinline__ int a_off(int v) { return v < 4 ? v : v + 4; }

// Wave-relative LDS byte address from a generic (flat) shared pointer:
// per CDNA5 ISA §10.2, LDS aperture addresses use addr[31:0] directly.
__device__ __forceinline__ unsigned lds_addr(const void* p) {
  return (unsigned)(unsigned long long)p;
}

// Async global->LDS copy of 16 bytes per lane (GVS addressing), ASYNCcnt-tracked.
__device__ __forceinline__ void async_copy_b128(unsigned lds_byte_addr,
                                                unsigned gvoff_bytes,
                                                const void* sbase) {
  asm volatile("global_load_async_to_lds_b128 %0, %1, %2"
               :: "v"(lds_byte_addr), "v"(gvoff_bytes), "s"(sbase)
               : "memory");
}

// ---------------------------------------------------------------------------
// Kernel 1: fused stream->head weight-gather + QKV projection.
//   Q[bh,t,d] (bf16, pre-scaled by 1/sqrt(D)), K[bh,t,d] (bf16),
//   Vt[bh,d,t] (bf16, transposed for the PV GEMM B-fragment layout).
// ---------------------------------------------------------------------------
__global__ __launch_bounds__(256)
void qkv_proj_kernel(const float* __restrict__ x,
                     const float* __restrict__ Wqkv,
                     const int*   __restrict__ perm,
                     unsigned short* __restrict__ Qb,
                     unsigned short* __restrict__ Kb,
                     unsigned short* __restrict__ Vt)
{
  __shared__ __align__(16) unsigned short xs [QTILE * D_];   // 128x64 bf16, 16KB
  __shared__ __align__(16) unsigned short wsT[OD_ * D_];     // 192x64 bf16 (W^T), 24KB

  const int bh    = blockIdx.x;           // b*H + h
  const int tbase = blockIdx.y * QTILE;
  const int b = bh >> 5;
  const int h = bh & 31;
  const int n = h >> 2;                   // h = n*K + k
  const int k = h & 3;

  const int p = perm[k * N_ + n];         // inv∘perm folds: only weight gather survives
  const float* Wsrc = Wqkv + (size_t)(k * N_ + p) * D_ * OD_;

  const int tid = threadIdx.x;

  // Stage x tile (fp32 -> bf16)
  for (int i = tid; i < QTILE * D_; i += 256) {
    int row = i >> 6, col = i & 63;
    float v = x[(size_t)(((b * T_ + tbase + row) * N_ + n) * K_ + k) * D_ + col];
    xs[i] = f2bf(v);
  }
  // Stage W^T (fp32 -> bf16): wsT[o][d] = W[d][o] so B-frag K-pairs are contiguous
  for (int i = tid; i < D_ * OD_; i += 256) {
    int d = i / OD_, o = i % OD_;
    wsT[o * D_ + d] = f2bf(Wsrc[i]);
  }
  __syncthreads();

  const unsigned int* xs_u = (const unsigned int*)xs;
  const unsigned int* wt_u = (const unsigned int*)wsT;

  const int wv   = tid >> 5;
  const int lane = tid & 31;
  const int half = lane >> 4;
  const int l16  = lane & 15;

  v8f acc[12] = {};

  for (int ks = 0; ks < 2; ++ks) {        // K-dim = 64 = 2 x 32
    v8u au;
    const int arow = (wv * 16 + l16) * 32 + ks * 16 + 8 * half;
#pragma unroll
    for (int v = 0; v < 8; ++v) au[v] = xs_u[arow + v];
    const v16bf afrag = __builtin_bit_cast(v16bf, au);
#pragma unroll
    for (int ct = 0; ct < 12; ++ct) {     // 12 x 16 = 192 output cols
      v8u bu;
      const int brow = (ct * 16 + l16) * 32 + ks * 16 + 8 * half;
#pragma unroll
      for (int v = 0; v < 8; ++v) bu[v] = wt_u[brow + v];
      const v16bf bfrag = __builtin_bit_cast(v16bf, bu);
      acc[ct] = __builtin_amdgcn_wmma_f32_16x16x32_bf16(
          false, afrag, false, bfrag, (short)0, acc[ct], false, false);
    }
  }

  // Epilogue: ct 0-3 -> Q, 4-7 -> K, 8-11 -> V^T
#pragma unroll
  for (int ct = 0; ct < 12; ++ct) {
    const int part = ct >> 2;
    const int d    = (ct & 3) * 16 + l16;
#pragma unroll
    for (int r = 0; r < 8; ++r) {
      const int t = tbase + wv * 16 + r + 8 * half;     // C layout: M = r + 8*half
      const float val = acc[ct][r];
      if (part == 0)      Qb[(size_t)(bh * T_ + t) * D_ + d] = f2bf(val * 0.125f);
      else if (part == 1) Kb[(size_t)(bh * T_ + t) * D_ + d] = f2bf(val);
      else                Vt[(size_t)(bh * D_ + d) * T_ + t] = f2bf(val);
    }
  }
}

// ---------------------------------------------------------------------------
// Kernel 2: causal flash attention, one (b,head) x 128 queries per block,
// 16 queries per wave, 64-key blocks staged to LDS via async-to-LDS copies,
// online softmax, bf16 WMMA + f32 accumulation.
// ---------------------------------------------------------------------------
__global__ __launch_bounds__(256)
void flash_attn_kernel(const unsigned short* __restrict__ Qb,
                       const unsigned short* __restrict__ Kb,
                       const unsigned short* __restrict__ Vt,
                       float* __restrict__ out)
{
  __shared__ __align__(16) unsigned short Ks[64 * 64];        // K block  [t][d], 8KB
  __shared__ __align__(16) unsigned short Vs[64 * 64];        // V block  [d][t], 8KB
  __shared__ __align__(16) unsigned short Plds[8 * 16 * 64];  // per-wave P tile, 16KB

  const int bh    = blockIdx.x;
  const int qbase = blockIdx.y * QTILE;
  const int b = bh >> 5, h = bh & 31, n = h >> 2, k = h & 3;

  const int tid  = threadIdx.x;
  const int wv   = tid >> 5;
  const int lane = tid & 31;
  const int half = lane >> 4;
  const int l16  = lane & 15;
  const int q0   = qbase + wv * 16;

  const unsigned int* Qu  = (const unsigned int*)Qb;
  const unsigned int* Ksu = (const unsigned int*)Ks;
  const unsigned int* Vsu = (const unsigned int*)Vs;
  const unsigned int* Pu  = (const unsigned int*)Plds;

  // Uniform global bases for this (b,head); async copies use GVS addressing.
  const unsigned short* KbBH = Kb + (size_t)bh * T_ * D_;     // [t][d]
  const unsigned short* VtBH = Vt + (size_t)bh * D_ * T_;     // [d][t]

  // Per-thread staging chunks: 2 x 16B for K, 2 x 16B for V per lane.
  // chunk c in [0,512): row = c>>3, col = (c&7)*8 (8 bf16 = 16B).
  const int c0 = tid, c1 = tid + 256;
  const int r0 = c0 >> 3, o0 = (c0 & 7) * 8;
  const int r1 = c1 >> 3, o1 = (c1 & 7) * 8;
  const unsigned ksA0 = lds_addr(&Ks[r0 * 64 + o0]);
  const unsigned ksA1 = lds_addr(&Ks[r1 * 64 + o1]);
  const unsigned vsA0 = lds_addr(&Vs[r0 * 64 + o0]);
  const unsigned vsA1 = lds_addr(&Vs[r1 * 64 + o1]);

  // Load Q A-fragments once (d = 2 k-steps of 32)
  v16bf qfrag[2];
  {
    const int rowbase = (bh * T_ + (q0 + l16)) * 32;
#pragma unroll
    for (int ks = 0; ks < 2; ++ks) {
      v8u au;
#pragma unroll
      for (int v = 0; v < 8; ++v) au[v] = Qu[rowbase + ks * 16 + 8 * half + v];
      qfrag[ks] = __builtin_bit_cast(v16bf, au);
    }
  }

  float mrow[8], lrow[8];
  v8f accO[4] = {};
#pragma unroll
  for (int r = 0; r < 8; ++r) { mrow[r] = -1e30f; lrow[r] = 0.0f; }

  const int kend = qbase + QTILE;                 // causal: keys < qbase+128
  for (int kb = 0; kb < kend; kb += 64) {
    // ---- async-stage K block [kb..kb+64)x[0..64) and V^T block [0..64)x[kb..kb+64) ----
    async_copy_b128(ksA0, (unsigned)(((kb + r0) * D_ + o0) * 2), KbBH);
    async_copy_b128(ksA1, (unsigned)(((kb + r1) * D_ + o1) * 2), KbBH);
    async_copy_b128(vsA0, (unsigned)((r0 * T_ + kb + o0) * 2), VtBH);
    async_copy_b128(vsA1, (unsigned)((r1 * T_ + kb + o1) * 2), VtBH);
    asm volatile("s_wait_asynccnt 0" ::: "memory");
    __syncthreads();

    // ---- S = Q K^T (16 x 64) ----
    v8f accS[4] = {};
#pragma unroll
    for (int ks = 0; ks < 2; ++ks) {
#pragma unroll
      for (int ct = 0; ct < 4; ++ct) {
        const int base = (ct * 16 + l16) * 32 + ks * 16 + 8 * half;
        v8u bu;
#pragma unroll
        for (int v = 0; v < 8; ++v) bu[v] = Ksu[base + v];
        const v16bf kf = __builtin_bit_cast(v16bf, bu);
        accS[ct] = __builtin_amdgcn_wmma_f32_16x16x32_bf16(
            false, qfrag[ks], false, kf, (short)0, accS[ct], false, false);
      }
    }

    // ---- causal mask + online softmax (row = (r, half); cols across 16 lanes) ----
#pragma unroll
    for (int r = 0; r < 8; ++r) {
      const int qg = q0 + r + 8 * half;
      float mloc = -1e30f;
#pragma unroll
      for (int ct = 0; ct < 4; ++ct) {
        const int col = kb + ct * 16 + l16;
        const float s = (col <= qg) ? accS[ct][r] : -1e30f;
        accS[ct][r] = s;
        mloc = fmaxf(mloc, s);
      }
#pragma unroll
      for (int off = 1; off < 16; off <<= 1)
        mloc = fmaxf(mloc, __shfl_xor(mloc, off, 32));
      const float mnew  = fmaxf(mrow[r], mloc);
      const float scale = __expf(mrow[r] - mnew);
      float psum = 0.0f;
#pragma unroll
      for (int ct = 0; ct < 4; ++ct) {
        const float pv = __expf(accS[ct][r] - mnew);
        accS[ct][r] = pv;
        psum += pv;
      }
#pragma unroll
      for (int off = 1; off < 16; off <<= 1)
        psum += __shfl_xor(psum, off, 32);
      lrow[r] = lrow[r] * scale + psum;
      mrow[r] = mnew;
#pragma unroll
      for (int dt = 0; dt < 4; ++dt) accO[dt][r] *= scale;
    }

    // ---- P: C-layout -> A-layout via this wave's private LDS tile ----
#pragma unroll
    for (int ct = 0; ct < 4; ++ct)
#pragma unroll
      for (int r = 0; r < 8; ++r)
        Plds[wv * 1024 + (r + 8 * half) * 64 + ct * 16 + l16] = f2bf(accS[ct][r]);
    // same-wave DS ops are in-order; wait keeps the compiler from reordering
    asm volatile("s_wait_dscnt 0" ::: "memory");

    // ---- O += P @ V ----
#pragma unroll
    for (int ks = 0; ks < 2; ++ks) {
      v8u au;
      const int prow = wv * 512 + l16 * 32 + ks * 16 + 8 * half;
#pragma unroll
      for (int v = 0; v < 8; ++v) au[v] = Pu[prow + v];
      const v16bf pf = __builtin_bit_cast(v16bf, au);
#pragma unroll
      for (int dt = 0; dt < 4; ++dt) {
        const int base = (dt * 16 + l16) * 32 + ks * 16 + 8 * half;
        v8u bu;
#pragma unroll
        for (int v = 0; v < 8; ++v) bu[v] = Vsu[base + v];
        const v16bf vf = __builtin_bit_cast(v16bf, bu);
        accO[dt] = __builtin_amdgcn_wmma_f32_16x16x32_bf16(
            false, pf, false, vf, (short)0, accO[dt], false, false);
      }
    }
    __syncthreads();    // staging buffers reused next iteration
  }

  // ---- epilogue: out[b,t,n,k,d] fp32 (stream-order output) ----
#pragma unroll
  for (int r = 0; r < 8; ++r) {
    const float inv_l = 1.0f / lrow[r];
    const int t = q0 + r + 8 * half;
    const size_t ob = ((size_t)((b * T_ + t) * N_ + n) * K_ + k) * D_;
#pragma unroll
    for (int dt = 0; dt < 4; ++dt)
      out[ob + dt * 16 + l16] = accO[dt][r] * inv_l;
  }
}

// ---------------------------------------------------------------------------
extern "C" void kernel_launch(void* const* d_in, const int* in_sizes, int n_in,
                              void* d_out, int out_size, void* d_ws, size_t ws_size,
                              hipStream_t stream) {
  const float* x    = (const float*)d_in[0];   // [B,T,N,K,D] fp32
  const float* Wqkv = (const float*)d_in[1];   // [K,N,D,3D]  fp32
  const int*   perm = (const int*)d_in[2];     // [K,N] int32
  // d_in[3] (inv_kn) is intentionally unused: inv∘perm composes to identity
  // on the data path; only the weight gather survives.
  float* out = (float*)d_out;

  const size_t per = (size_t)B_ * H_ * T_ * D_;          // 8.4M bf16 elems each
  unsigned short* Qb = (unsigned short*)d_ws;
  unsigned short* Kb = Qb + per;
  unsigned short* Vt = Kb + per;                          // total 48MB of d_ws

  dim3 grid(B_ * H_, T_ / QTILE);                         // (128, 8)
  qkv_proj_kernel<<<grid, 256, 0, stream>>>(x, Wqkv, perm, Qb, Kb, Vt);
  flash_attn_kernel<<<grid, 256, 0, stream>>>(Qb, Kb, Vt, out);
}
